// GlobalGraph_83872121356512
// MI455X (gfx1250) — compile-verified, matrix operand order checked by reading, not run on hardware
//
#include <hip/hip_runtime.h>
#include <hip/hip_bf16.h>

// CDNA5 / gfx1250: wave32, WMMA 16x16x32 bf16 -> f32 accum.
typedef __attribute__((ext_vector_type(16))) __bf16 v16bf;
typedef __attribute__((ext_vector_type(8)))  float  v8f;

#define HID   1024
#define HEADS 16
#define HD    64
#define BB    4
#define SS    2048
#define NTOK  (BB*SS)          // 8192
#define NCOL  (3*HID)          // fused q|k|v output columns = 3072

// ---------------------------------------------------------------------------
// Kernel 1: fp32 -> bf16 conversion.
//   Xbf  [NTOK][HID]   bf16 row-major activations
//   Wt   [NCOL][HID]   bf16, W^T fused (q rows 0..1023, k 1024.., v 2048..)
// ---------------------------------------------------------------------------
__global__ void convert_kernel(const float* __restrict__ X,
                               const float* __restrict__ Wq,
                               const float* __restrict__ Wk,
                               const float* __restrict__ Wv,
                               __bf16* __restrict__ Xbf,
                               __bf16* __restrict__ Wt) {
  int idx = blockIdx.x * blockDim.x + threadIdx.x;
  if (idx < NTOK * HID) {
    Xbf[idx] = (__bf16)X[idx];
  }
  if (idx < 3 * HID * HID) {
    int w = idx >> 20;               // which matrix
    int r = idx & (HID * HID - 1);   // r = n*HID + k in output layout
    int n = r >> 10;
    int k = r & (HID - 1);
    const float* W = (w == 0) ? Wq : ((w == 1) ? Wk : Wv);
    Wt[idx] = (__bf16)W[k * HID + n];   // transpose: Wt[n][k] = W[k][n]
  }
}

// ---------------------------------------------------------------------------
// Kernel 2: fused QKV GEMM.  One wave = one 16(M) x 64(N) tile of
// C[8192 x 3072]: per k-step, ONE A fragment is reused across FOUR B
// fragments -> 4 WMMAs per A load.  128 WMMAs per wave over K=1024.
// Outputs (bf16):
//   Q [B,H,S,D]  pre-scaled by 1/sqrt(D)
//   K [B,H,S,D]
//   Vt[B,H,D,S]  (transposed so PV B-fragments are contiguous over keys)
// ---------------------------------------------------------------------------
__global__ void qkv_gemm_kernel(const __bf16* __restrict__ Xbf,
                                const __bf16* __restrict__ Wt,
                                const float* __restrict__ bq,
                                const float* __restrict__ bk,
                                const float* __restrict__ bv,
                                __bf16* __restrict__ Qbf,
                                __bf16* __restrict__ Kbf,
                                __bf16* __restrict__ Vt) {
  int gwave = (blockIdx.x * blockDim.x + threadIdx.x) >> 5;  // global wave id
  int lane  = threadIdx.x & 31;
  int l16   = lane & 15;
  int half  = lane >> 4;

  int tileM = gwave & 511;   // 512 row tiles (8192/16)
  int tileN = gwave >> 9;    // 48 col tiles, each 64 wide (3072/64)

  const __bf16* ap  = Xbf + (tileM * 16 + l16) * HID + half * 16;
  const __bf16* bp0 = Wt + (tileN * 64 +  0 + l16) * HID + half * 16;
  const __bf16* bp1 = Wt + (tileN * 64 + 16 + l16) * HID + half * 16;
  const __bf16* bp2 = Wt + (tileN * 64 + 32 + l16) * HID + half * 16;
  const __bf16* bp3 = Wt + (tileN * 64 + 48 + l16) * HID + half * 16;

  v8f acc0 = {}, acc1 = {}, acc2 = {}, acc3 = {};
#pragma unroll 2
  for (int k = 0; k < HID; k += 32) {
    __builtin_prefetch(ap + k + 256, 0, 0);
    v16bf a  = *(const v16bf*)(ap  + k);
    v16bf b0 = *(const v16bf*)(bp0 + k);
    v16bf b1 = *(const v16bf*)(bp1 + k);
    v16bf b2 = *(const v16bf*)(bp2 + k);
    v16bf b3 = *(const v16bf*)(bp3 + k);
    acc0 = __builtin_amdgcn_wmma_f32_16x16x32_bf16(false, a, false, b0,
                                                   (short)0, acc0, false, false);
    acc1 = __builtin_amdgcn_wmma_f32_16x16x32_bf16(false, a, false, b1,
                                                   (short)0, acc1, false, false);
    acc2 = __builtin_amdgcn_wmma_f32_16x16x32_bf16(false, a, false, b2,
                                                   (short)0, acc2, false, false);
    acc3 = __builtin_amdgcn_wmma_f32_16x16x32_bf16(false, a, false, b3,
                                                   (short)0, acc3, false, false);
  }

  int which = tileN >> 4;        // 0=q, 1=k, 2=v  (16 tiles of 64 per matrix)
  int h     = tileN & 15;        // head index (64-aligned columns)
  const float* bias_p = (which == 0) ? bq : ((which == 1) ? bk : bv);
  float scale = (which == 0) ? 0.125f : 1.0f;   // fold 1/sqrt(64) into Q

#pragma unroll
  for (int g = 0; g < 4; ++g) {
    v8f accv = (g == 0) ? acc0 : (g == 1) ? acc1 : (g == 2) ? acc2 : acc3;
    int d    = g * 16 + l16;
    float bias = bias_p[h * HD + d];
#pragma unroll
    for (int i = 0; i < 8; ++i) {
      int t  = tileM * 16 + half * 8 + i;    // token index
      int bi = t >> 11;
      int s  = t & (SS - 1);
      int bh = bi * HEADS + h;
      float v = (accv[i] + bias) * scale;
      if (which == 2) {
        Vt[(bh * HD + d) * SS + s] = (__bf16)v;
      } else {
        __bf16* dst = (which == 0) ? Qbf : Kbf;
        dst[(bh * SS + s) * HD + d] = (__bf16)v;
      }
    }
  }
}

// ---------------------------------------------------------------------------
// Kernel 3: flash attention, one wave per block (barriers -> S_NOP, only the
// s_wait_dscnt survives).  Per 32-key block:
//   - 4 score WMMAs (2 subtiles x D split 64=2x32)
//   - ONE joint max update (shuffle-max only; halves shuffle traffic)
//   - P staged to LDS (C-layout -> A-layout)
//   - row-sums ell computed by WMMA(P, ones) on the matrix pipe instead of
//     64 shuffle-adds on the VALU pipe (WMMA co-executes with VALU)
//   - 4 PV WMMAs
// ---------------------------------------------------------------------------
__global__ void __launch_bounds__(32)
attn_kernel(const __bf16* __restrict__ Q,
            const __bf16* __restrict__ K,
            const __bf16* __restrict__ Vt,
            const float* __restrict__ mask,
            float* __restrict__ out) {
  __shared__ __attribute__((aligned(32))) __bf16 Plds[16][32];

  int lane = threadIdx.x & 31;
  int l16  = lane & 15;
  int half = lane >> 4;

  int job = blockIdx.x;              // 8192 jobs = 64 (b,h) * 128 q-tiles
  int bh  = job >> 7;
  int qt  = job & 127;
  int b   = bh >> 4;
  int h   = bh & 15;

  // Q A-fragments: lane -> query row, two 32-wide K-dim slices of D=64.
  const __bf16* qbase = Q + (bh * SS + qt * 16 + l16) * HD;
  v16bf qa0 = *(const v16bf*)(qbase + half * 16);        // d 0..31
  v16bf qa1 = *(const v16bf*)(qbase + 32 + half * 16);   // d 32..63

  // All-ones B fragment: WMMA(P, ones) -> row sums of P in C layout.
  v16bf ones;
#pragma unroll
  for (int j = 0; j < 16; ++j) ones[j] = (__bf16)1.0f;

  v8f acc0 = {}, acc1 = {}, acc2 = {}, acc3 = {};
  float mrow[8], lrow[8];
#pragma unroll
  for (int i = 0; i < 8; ++i) { mrow[i] = -3.0e38f; lrow[i] = 0.0f; }

  for (int kb = 0; kb < SS; kb += 32) {
    // ---- scores for both 16-key subtiles -------------------------------
    v8f sc0, sc1;
#pragma unroll
    for (int sub = 0; sub < 2; ++sub) {
      int kidx = kb + sub * 16 + l16;   // key column handled by this lane
      const __bf16* kbase = K + (bh * SS + kidx) * HD;
      v16bf kf0 = *(const v16bf*)(kbase + half * 16);
      v16bf kf1 = *(const v16bf*)(kbase + 32 + half * 16);
      v8f sc = {};
      sc = __builtin_amdgcn_wmma_f32_16x16x32_bf16(false, qa0, false, kf0,
                                                   (short)0, sc, false, false);
      sc = __builtin_amdgcn_wmma_f32_16x16x32_bf16(false, qa1, false, kf1,
                                                   (short)0, sc, false, false);
      float mbias = (1.0f - mask[b * SS + kidx]) * -10000.0f;
#pragma unroll
      for (int i = 0; i < 8; ++i) sc[i] += mbias;
      if (sub == 0) sc0 = sc; else sc1 = sc;
    }

    // ---- single online-softmax max update for the whole 32-key block ----
#pragma unroll
    for (int i = 0; i < 8; ++i) {
      float mx = fmaxf(sc0[i], sc1[i]);
      mx = fmaxf(mx, __shfl_xor(mx, 8, 16));
      mx = fmaxf(mx, __shfl_xor(mx, 4, 16));
      mx = fmaxf(mx, __shfl_xor(mx, 2, 16));
      mx = fmaxf(mx, __shfl_xor(mx, 1, 16));
      float mnew = fmaxf(mrow[i], mx);
      float corr = __expf(mrow[i] - mnew);
      mrow[i] = mnew;
      acc0[i] *= corr; acc1[i] *= corr; acc2[i] *= corr; acc3[i] *= corr;
      lrow[i] *= corr;
      // stage P: row = half*8+i, cols = lane column in each subtile
      Plds[half * 8 + i][     l16] = (__bf16)__expf(sc0[i] - mnew);
      Plds[half * 8 + i][16 + l16] = (__bf16)__expf(sc1[i] - mnew);
    }
    __syncthreads();   // single-wave WG: barrier = S_NOP, keeps s_wait_dscnt

    // P as A-fragment: lane -> query row, 16 contiguous keys per lane half.
    v16bf pa = *(const v16bf*)(&Plds[l16][half * 16]);

    // ell row-sums on the matrix pipe: every column of WMMA(P, ones) holds
    // the row sum, in the same C-layout [i] indexing as the ctx accumulators.
    v8f zero = {};
    v8f sums = __builtin_amdgcn_wmma_f32_16x16x32_bf16(false, pa, false, ones,
                                                       (short)0, zero, false, false);
#pragma unroll
    for (int i = 0; i < 8; ++i) lrow[i] += sums[i];

#pragma unroll
    for (int g = 0; g < 4; ++g) {
      // Vt B-fragment: lane -> output dim column, contiguous over keys.
      const __bf16* vb = Vt + (bh * HD + g * 16 + l16) * SS + kb + half * 16;
      v16bf vf = *(const v16bf*)vb;
      v8f* pacc = (g == 0) ? &acc0 : (g == 1) ? &acc1 : (g == 2) ? &acc2 : &acc3;
      *pacc = __builtin_amdgcn_wmma_f32_16x16x32_bf16(false, pa, false, vf,
                                                      (short)0, *pacc, false, false);
    }
    __syncthreads();   // protect Plds before next block overwrites
  }

  // Normalize (one reciprocal per row) and store fp32 context [B,S,H*D].
#pragma unroll
  for (int i = 0; i < 8; ++i) {
    float inv  = 1.0f / lrow[i];
    int   srow = qt * 16 + half * 8 + i;
    float* orow = out + (b * SS + srow) * HID + h * HD + l16;
    orow[ 0] = acc0[i] * inv;
    orow[16] = acc1[i] * inv;
    orow[32] = acc2[i] * inv;
    orow[48] = acc3[i] * inv;
  }
}

// ---------------------------------------------------------------------------
extern "C" void kernel_launch(void* const* d_in, const int* in_sizes, int n_in,
                              void* d_out, int out_size, void* d_ws, size_t ws_size,
                              hipStream_t stream) {
  const float* hs   = (const float*)d_in[0];
  const float* mask = (const float*)d_in[1];
  const float* Wq   = (const float*)d_in[2];
  const float* bq   = (const float*)d_in[3];
  const float* Wk   = (const float*)d_in[4];
  const float* bk   = (const float*)d_in[5];
  const float* Wv   = (const float*)d_in[6];
  const float* bv   = (const float*)d_in[7];
  float* out = (float*)d_out;

  char* ws = (char*)d_ws;
  __bf16* Xbf = (__bf16*)(ws);                                   // 16 MiB
  __bf16* Wt  = (__bf16*)(ws + (size_t)16777216);                //  6 MiB
  __bf16* Qbf = (__bf16*)(ws + (size_t)16777216 + 6291456);      // 16 MiB
  __bf16* Kbf = Qbf + (size_t)NTOK * HID;                        // 16 MiB
  __bf16* Vt  = Kbf + (size_t)NTOK * HID;                        // 16 MiB

  // 1) fp32 -> bf16 (+ fused transposed weights)
  convert_kernel<<<(NTOK * HID) / 256, 256, 0, stream>>>(hs, Wq, Wk, Wv, Xbf, Wt);

  // 2) fused QKV GEMM: 512*48 = 24576 tile-waves, 8 waves per 256-thread block
  qkv_gemm_kernel<<<24576 / 8, 256, 0, stream>>>(Xbf, Wt, bq, bk, bv, Qbf, Kbf, Vt);

  // 3) flash attention: 8192 wave-jobs, one wave per block
  attn_kernel<<<8192, 32, 0, stream>>>(Qbf, Kbf, Vt, mask, out);
}
